// FeedbackTransformerModel_86019605004854
// MI455X (gfx1250) — matrix operand order repaired and checked by reading, not compiled, not run on hardware
//
#include <hip/hip_runtime.h>
#include <cstdint>
#include <cstddef>

// ---------------------------------------------------------------------------
// Model constants (from reference)
// ---------------------------------------------------------------------------
constexpr int B = 4, S = 512, D = 512, H = 8, HD = 64, L = 2, FF = 2048, V = 32000;
constexpr float ATT_SCALE = 0.125f;                  // 1/sqrt(64)
constexpr float EMB_SCALE = 22.62741699796952f;      // sqrt(512)

typedef __attribute__((ext_vector_type(16))) __bf16 v16bf;
typedef __attribute__((ext_vector_type(8)))  float  v8f;
typedef unsigned int u32x4 __attribute__((ext_vector_type(4)));
typedef int          i32x4 __attribute__((ext_vector_type(4)));
typedef int          i32x8 __attribute__((ext_vector_type(8)));

#define DEVFN __device__ __forceinline__

// ---------------------------------------------------------------------------
// bf16 helpers (round-to-nearest-even)
// ---------------------------------------------------------------------------
DEVFN unsigned short f2bfbits(float f) {
  unsigned u = __builtin_bit_cast(unsigned, f);
  unsigned r = u + 0x7fffu + ((u >> 16) & 1u);
  return (unsigned short)(r >> 16);
}

DEVFN float wred_sum(float v) {
#pragma unroll
  for (int o = 16; o; o >>= 1) v += __shfl_xor(v, o, 32);
  return v;
}
DEVFN float wred_max(float v) {
#pragma unroll
  for (int o = 16; o; o >>= 1) v = fmaxf(v, __shfl_xor(v, o, 32));
  return v;
}

// ---------------------------------------------------------------------------
// WMMA A-fragment from bf16 source (row-major; pointer pre-offset by
// row*lda + (lane>>4)*16).  Contiguous K order: elems0-15 = K k0..k0+15
// (matches B fragment).  Two aligned 16B loads, zero ALU.
// ---------------------------------------------------------------------------
DEVFN v16bf load_afrag(const unsigned short* a0) {
  uint4 lo = *(const uint4*)(a0);
  uint4 hh = *(const uint4*)(a0 + 8);
  v16bf af;
  __builtin_memcpy(&af, &lo, 16);
  __builtin_memcpy((char*)&af + 16, &hh, 16);
  return af;
}

// ---------------------------------------------------------------------------
// Workgroup GEMM: Y(rows<B x N) = act( A(16 x K) @ Wt^T + bias )
//  - A: bf16 (LDS or global), row stride lda
//  - Wt: bf16 transposed weights, row n holds K contiguous values
//  - all 32 waves of a 1024-thread WG; wave w owns N-tiles w, w+32, ...
//  - optional f32 output Y and/or bf16 mirror Ybf (pad rows never stored)
//  - k-loop capped at unroll 2 to keep B-fragments from spilling to scratch
// ---------------------------------------------------------------------------
__device__ void wg_gemm(const unsigned short* A, int lda,
                        const unsigned short* Wt, const float* bias,
                        float* Y, long long ldy, unsigned short* Ybf,
                        int N, int K, bool relu, int lane, int wave) {
  const int mrow = lane & 15;
  const int hi   = lane >> 4;
  const unsigned short* arow = A + (long long)mrow * lda + hi * 16;
  for (int nt = wave; nt < (N >> 4); nt += 32) {
    const int n = (nt << 4) + (lane & 15);
    const unsigned short* brow = Wt + (long long)n * K + hi * 16;
    v8f acc = {0.f, 0.f, 0.f, 0.f, 0.f, 0.f, 0.f, 0.f};
#pragma unroll 2
    for (int kt = 0; kt < (K >> 5); ++kt) {
      v16bf af = load_afrag(arow + kt * 32);
      v16bf bf = *(const v16bf*)(brow + kt * 32);
      acc = __builtin_amdgcn_wmma_f32_16x16x32_bf16(false, af, false, bf,
                                                    (short)0, acc, false, false);
    }
    const float bv = bias ? bias[n] : 0.f;
#pragma unroll
    for (int r = 0; r < 8; ++r) {
      int mm = r + (hi << 3);
      if (mm < B) {
        float v = acc[r] + bv;
        if (relu) v = fmaxf(v, 0.f);
        if (Y)   Y[(long long)mm * ldy + n] = v;
        if (Ybf) Ybf[(long long)mm * ldy + n] = f2bfbits(v);
      }
    }
  }
}

// ---------------------------------------------------------------------------
// Kernel: convert + transpose f32 weight (K x N) -> bf16 (N x K)
// ---------------------------------------------------------------------------
__global__ void k_convert_transpose(const float* __restrict__ W,
                                    unsigned short* __restrict__ Wt,
                                    int K, int N) {
  long long total = (long long)K * N;
  for (long long i = (long long)blockIdx.x * blockDim.x + threadIdx.x; i < total;
       i += (long long)gridDim.x * blockDim.x) {
    int k = (int)(i / N), n = (int)(i % N);
    Wt[(long long)n * K + k] = f2bfbits(W[i]);
  }
}

// ---------------------------------------------------------------------------
// Kernel: embedding  emb[b,s,:] = tok_emb[id]*EMB_SCALE + pos_emb[s]
// ---------------------------------------------------------------------------
__global__ void k_embed(const int* __restrict__ ids, const float* __restrict__ tok,
                        const float* __restrict__ pos, float* __restrict__ emb) {
  int bs = blockIdx.x;                // b*S + s
  int s = bs & (S - 1);
  int id = ids[bs];
  const float* te = tok + (long long)id * D;
  const float* pe = pos + (long long)s * D;
  float* o = emb + (long long)bs * D;
  for (int d = threadIdx.x; d < D; d += blockDim.x)
    o[d] = te[d] * EMB_SCALE + pe[d];
}

// ---------------------------------------------------------------------------
// Sequential decoder: single persistent workgroup (32 wave32 waves),
// loops t = 0..S-1 carrying the KV memory recurrence.
// ---------------------------------------------------------------------------
struct SeqArgs {
  const float *emb, *lw;
  const float *kb, *vb, *qb, *ob, *f1b, *f2b;
  const float *ln1w, *ln1b, *ln2w, *ln2b;
  const unsigned short *kWt, *vWt, *qWt, *oWt, *f1Wt, *f2Wt;
  float *memk, *memv, *hidden;
  float *qbuf, *attp, *ffp, *macc, *sbuf;
  unsigned short *attbh, *ffbh, *maccbh;
};

__global__ void __launch_bounds__(1024) k_seq(SeqArgs a) {
  __shared__ float          xb [16][D];   // 32 KB: current x (f32)
  __shared__ unsigned short xbh[16][D];   // 16 KB: bf16 mirror (WMMA A source)
  const int tid  = threadIdx.x;
  const int lane = tid & 31;
  const int wave = tid >> 5;

  for (int i = tid; i < 16 * D; i += 1024) { (&xb[0][0])[i] = 0.f; (&xbh[0][0])[i] = 0; }

  // softmax(layer_weights) -- 3 scalars, computed per-thread
  float w0, w1, w2;
  {
    float l0 = a.lw[0], l1 = a.lw[1], l2 = a.lw[2];
    float mx = fmaxf(l0, fmaxf(l1, l2));
    float e0 = __expf(l0 - mx), e1 = __expf(l1 - mx), e2 = __expf(l2 - mx);
    float rs = 1.f / (e0 + e1 + e2);
    w0 = e0 * rs; w1 = e1 * rs; w2 = e2 * rs;
  }
  __syncthreads();

  for (int t = 0; t < S; ++t) {
    // ---- load x = emb[:, t] into LDS (f32 + bf16 mirror)
    for (int i = tid; i < B * D; i += 1024) {
      int b = i >> 9, d = i & (D - 1);
      float v = a.emb[((long long)b * S + t) * D + d];
      xb[b][d] = v; xbh[b][d] = f2bfbits(v);
    }
    __syncthreads();

    if (t == 0) {  // bootstrap memory row 0 from x0 itself
      wg_gemm(&xbh[0][0], D, a.kWt, a.kb, a.memk, (long long)S * D, nullptr, D, D, false, lane, wave);
      wg_gemm(&xbh[0][0], D, a.vWt, a.vb, a.memv, (long long)S * D, nullptr, D, D, false, lane, wave);
      __threadfence(); __syncthreads();
    }
    const int nvalid = (t == 0) ? 1 : t;

    // ---- macc = w[0] * x   (weighted layer-output accumulator)
    for (int i = tid; i < B * D; i += 1024) {
      int b = i >> 9, d = i & (D - 1);
      a.macc[b * D + d] = w0 * xb[b][d];
    }
    __threadfence(); __syncthreads();

    for (int l = 0; l < L; ++l) {
      const float wl = (l == 0) ? w1 : w2;

      // Q projection (f32 out; attention consumes f32)
      wg_gemm(&xbh[0][0], D, a.qWt + (size_t)l * D * D, a.qb + l * D,
              a.qbuf, D, nullptr, D, D, false, lane, wave);
      __threadfence(); __syncthreads();

      // attention: one wave per (b, h)
      {
        const int b = wave >> 3, h = wave & 7;
        const float* qrow = a.qbuf + b * D + h * HD;
        float* srow = a.sbuf + (size_t)(b * H + h) * S;
        float mx = -1e30f;
        for (int m = lane; m < nvalid; m += 32) {
          const float* kr = a.memk + ((long long)b * S + m) * D + h * HD;
          float acc = 0.f;
          for (int j = 0; j < HD; ++j) acc = fmaf(qrow[j], kr[j], acc);
          acc *= ATT_SCALE;
          srow[m] = acc;
          mx = fmaxf(mx, acc);
        }
        mx = wred_max(mx);
        float den = 0.f;
        for (int m = lane; m < nvalid; m += 32) den += __expf(srow[m] - mx);
        den = wred_sum(den);
        float rden = 1.f / den;
        __threadfence(); __syncthreads();   // make srow visible across lanes
        for (int d = lane; d < HD; d += 32) {
          float acc = 0.f;
          for (int m = 0; m < nvalid; ++m) {
            float p = __expf(srow[m] - mx) * rden;
            acc = fmaf(p, a.memv[((long long)b * S + m) * D + h * HD + d], acc);
          }
          a.attbh[b * D + h * HD + d] = f2bfbits(acc);   // bf16 A for O-proj
        }
      }
      __threadfence(); __syncthreads();

      // output projection
      wg_gemm(a.attbh, D, a.oWt + (size_t)l * D * D, a.ob + l * D,
              a.attp, D, nullptr, D, D, false, lane, wave);
      __threadfence(); __syncthreads();

      // LN1 over (x + att): one wave per row
      if (wave < B) {
        float s = 0.f, ss = 0.f;
        for (int d = lane; d < D; d += 32) {
          float v = xb[wave][d] + a.attp[wave * D + d];
          s += v; ss += v * v;
        }
        s = wred_sum(s); ss = wred_sum(ss);
        float mean = s / D, var = ss / D - mean * mean;
        float rstd = rsqrtf(var + 1e-5f);
        for (int d = lane; d < D; d += 32) {
          float v = xb[wave][d] + a.attp[wave * D + d];
          float y = (v - mean) * rstd * a.ln1w[l * D + d] + a.ln1b[l * D + d];
          xb[wave][d] = y; xbh[wave][d] = f2bfbits(y);
        }
      }
      __syncthreads();

      // FF1 (relu, bf16-only output) and FF2
      wg_gemm(&xbh[0][0], D, a.f1Wt + (size_t)l * D * FF, a.f1b + l * FF,
              nullptr, FF, a.ffbh, FF, D, true, lane, wave);
      __threadfence(); __syncthreads();
      wg_gemm(a.ffbh, FF, a.f2Wt + (size_t)l * FF * D, a.f2b + l * D,
              a.ffp, D, nullptr, D, FF, false, lane, wave);
      __threadfence(); __syncthreads();

      // LN2 over (x + ff); accumulate weighted layer output (+ bf16 on last)
      if (wave < B) {
        float s = 0.f, ss = 0.f;
        for (int d = lane; d < D; d += 32) {
          float v = xb[wave][d] + a.ffp[wave * D + d];
          s += v; ss += v * v;
        }
        s = wred_sum(s); ss = wred_sum(ss);
        float mean = s / D, var = ss / D - mean * mean;
        float rstd = rsqrtf(var + 1e-5f);
        for (int d = lane; d < D; d += 32) {
          float v = xb[wave][d] + a.ffp[wave * D + d];
          float y = (v - mean) * rstd * a.ln2w[l * D + d] + a.ln2b[l * D + d];
          xb[wave][d] = y; xbh[wave][d] = f2bfbits(y);
          float mnew = a.macc[wave * D + d] + wl * y;
          a.macc[wave * D + d] = mnew;
          if (l == L - 1) a.maccbh[wave * D + d] = f2bfbits(mnew);
        }
      }
      __threadfence(); __syncthreads();
    }

    // ---- store hidden state for this step
    for (int i = tid; i < B * D; i += 1024) {
      int b = i >> 9, d = i & (D - 1);
      a.hidden[((long long)b * S + t) * D + d] = xb[b][d];
    }
    // ---- memory write: memk/v[:, t] = macc @ {kW,vW} + bias
    wg_gemm(a.maccbh, D, a.kWt, a.kb, a.memk + (long long)t * D, (long long)S * D, nullptr, D, D, false, lane, wave);
    wg_gemm(a.maccbh, D, a.vWt, a.vb, a.memv + (long long)t * D, (long long)S * D, nullptr, D, D, false, lane, wave);
    __threadfence(); __syncthreads();
  }
}

// ---------------------------------------------------------------------------
// Final LayerNorm -> bf16 activations (one block per row)
// ---------------------------------------------------------------------------
__global__ void k_lnf(const float* __restrict__ hidden, const float* __restrict__ w,
                      const float* __restrict__ b, unsigned short* __restrict__ out) {
  int row = blockIdx.x;
  const float* x = hidden + (long long)row * D;
  float s = 0.f, ss = 0.f;
  for (int d = threadIdx.x; d < D; d += blockDim.x) {
    float v = x[d]; s += v; ss += v * v;
  }
  __shared__ float sh[2][8];
  int lane = threadIdx.x & 31, wv = threadIdx.x >> 5;
  s = wred_sum(s); ss = wred_sum(ss);
  if (lane == 0) { sh[0][wv] = s; sh[1][wv] = ss; }
  __syncthreads();
  if (wv == 0) {
    float aa = (lane < 8) ? sh[0][lane] : 0.f;
    float bb = (lane < 8) ? sh[1][lane] : 0.f;
    aa = wred_sum(aa); bb = wred_sum(bb);
    if (lane == 0) { sh[0][0] = aa; sh[1][0] = bb; }
  }
  __syncthreads();
  float mean = sh[0][0] / D, var = sh[1][0] / D - mean * mean;
  float rstd = rsqrtf(var + 1e-5f);
  for (int d = threadIdx.x; d < D; d += blockDim.x)
    out[(long long)row * D + d] = f2bfbits((x[d] - mean) * rstd * w[d] + b[d]);
}

// ---------------------------------------------------------------------------
// Head projection: (2048 x 512) bf16 @ (512 x 32000) bf16 -> f32.
// Each block owns one 16-row A tile (16x512 bf16 = 16 KB), staged to LDS by
// the Tensor Data Mover (D# per ISA ch.8), then 8 waves compute 8 adjacent
// 16x16 N-tiles reusing it (ds_load_b128 fragments, global B fragments).
// k-loop capped at unroll 2 so B-fragments pipeline without scratch spills.
// ---------------------------------------------------------------------------
__global__ void __launch_bounds__(256) k_head(const unsigned short* __restrict__ hl,
                                              const unsigned short* __restrict__ Wt,
                                              float* __restrict__ out) {
  __shared__ unsigned short atile[16 * D];          // 16 KB, at LDS offset 0
  const int lane = threadIdx.x & 31;
  const int wv   = threadIdx.x >> 5;
  const int mt = blockIdx.x / (V / 16 / 8);         // 128 row tiles
  const int nc = blockIdx.x % (V / 16 / 8);         // 250 col chunks of 8 tiles
  const int nt = nc * 8 + wv;

  // ---- stage A tile into LDS
#if __has_builtin(__builtin_amdgcn_tensor_load_to_lds)
  if (threadIdx.x < 32) {                           // wave 0 issues the TDM op
    unsigned long long ga =
        (unsigned long long)(uintptr_t)(hl + (size_t)mt * 16 * D);
    // D# group0: count=1 | lds_addr=0 | global_addr | type=2
    u32x4 g0 = { 1u, 0u, (unsigned)(ga & 0xffffffffu),
                 (unsigned)((ga >> 32) & 0x01ffffffu) | (2u << 30) };
    // D# group1: data_size=2B; tensor 512x16; tile 512x16; dim0 stride 512
    i32x8 g1 = { (int)(1u << 16),     // [17:16] data_size=1 (2 bytes)
                 (int)(512u << 16),   // [63:48] tensor_dim0 = 512
                 (int)(16u << 16),    // [111:80] tensor_dim1 = 16
                 (int)(512u << 16),   // [127:112] tile_dim0 = 512
                 16,                  // [143:128] tile_dim1 = 16
                 512,                 // [207:160] tensor_dim0_stride = 512
                 0, 0 };
    i32x4 gz = { 0, 0, 0, 0 };
#if __has_include(<hip/amd_detail/amd_gfx1250_TDM.h>)
    i32x8 gz8 = { 0, 0, 0, 0, 0, 0, 0, 0 };
    __builtin_amdgcn_tensor_load_to_lds(g0, g1, gz, gz, gz8, 0);
#else
    __builtin_amdgcn_tensor_load_to_lds(g0, g1, gz, gz, 0);
#endif
    __builtin_amdgcn_s_wait_tensorcnt(0);
  }
#else
  {
    const uint4* src = (const uint4*)(hl + (size_t)mt * 16 * D);
    uint4* dst = (uint4*)atile;
    for (int i = threadIdx.x; i < 16 * D / 8; i += 256) dst[i] = src[i];
  }
#endif
  __syncthreads();

  // ---- one 16x16 tile per wave
  const int hi = lane >> 4;
  const unsigned short* arow = atile + (lane & 15) * D + hi * 16;   // LDS
  const unsigned short* brow = Wt + (long long)(nt * 16 + (lane & 15)) * D + hi * 16;
  __builtin_prefetch(brow, 0, 1);                   // warm GL2 for the B row
  v8f acc = {0.f, 0.f, 0.f, 0.f, 0.f, 0.f, 0.f, 0.f};
#pragma unroll 2
  for (int kt = 0; kt < D / 32; ++kt) {
    v16bf af = load_afrag(arow + kt * 32);
    v16bf bf = *(const v16bf*)(brow + kt * 32);
    acc = __builtin_amdgcn_wmma_f32_16x16x32_bf16(false, af, false, bf,
                                                  (short)0, acc, false, false);
  }
  const int m0 = mt * 16 + (hi << 3);
  const int n  = nt * 16 + (lane & 15);
#pragma unroll
  for (int r = 0; r < 8; ++r)
    out[(long long)(m0 + r) * V + n] = acc[r];
}

// ---------------------------------------------------------------------------
// Host entry
// ---------------------------------------------------------------------------
extern "C" void kernel_launch(void* const* d_in, const int* in_sizes, int n_in,
                              void* d_out, int out_size, void* d_ws, size_t ws_size,
                              hipStream_t stream) {
  (void)in_sizes; (void)n_in; (void)out_size; (void)ws_size;
  const int*   ids   = (const int*)  d_in[0];
  const float* tok   = (const float*)d_in[1];
  const float* pos   = (const float*)d_in[2];
  const float* lw    = (const float*)d_in[3];
  const float* kW    = (const float*)d_in[4];
  const float* kb    = (const float*)d_in[5];
  const float* vW    = (const float*)d_in[6];
  const float* vb    = (const float*)d_in[7];
  const float* qW    = (const float*)d_in[8];
  const float* qb    = (const float*)d_in[9];
  const float* oW    = (const float*)d_in[10];
  const float* ob    = (const float*)d_in[11];
  const float* f1W   = (const float*)d_in[12];
  const float* f1b   = (const float*)d_in[13];
  const float* f2W   = (const float*)d_in[14];
  const float* f2b   = (const float*)d_in[15];
  const float* ln1w  = (const float*)d_in[16];
  const float* ln1b  = (const float*)d_in[17];
  const float* ln2w  = (const float*)d_in[18];
  const float* ln2b  = (const float*)d_in[19];
  const float* lnfw  = (const float*)d_in[20];
  const float* lnfb  = (const float*)d_in[21];
  const float* headW = (const float*)d_in[22];
  float* out = (float*)d_out;

  // workspace carve-up (256B aligned)
  size_t off = 0;
  auto alloc = [&](size_t bytes) -> void* {
    off = (off + 255) & ~(size_t)255;
    void* p = (char*)d_ws + off;
    off += bytes;
    return p;
  };
  float* emb    = (float*)alloc((size_t)B * S * D * 4);
  float* hidden = (float*)alloc((size_t)B * S * D * 4);
  float* memk   = (float*)alloc((size_t)B * S * D * 4);
  float* memv   = (float*)alloc((size_t)B * S * D * 4);
  unsigned short* hl     = (unsigned short*)alloc((size_t)B * S * D * 2);
  unsigned short* kWt    = (unsigned short*)alloc((size_t)D * D * 2);
  unsigned short* vWt    = (unsigned short*)alloc((size_t)D * D * 2);
  unsigned short* qWt    = (unsigned short*)alloc((size_t)L * D * D * 2);
  unsigned short* oWt    = (unsigned short*)alloc((size_t)L * D * D * 2);
  unsigned short* f1Wt   = (unsigned short*)alloc((size_t)L * D * FF * 2);
  unsigned short* f2Wt   = (unsigned short*)alloc((size_t)L * FF * D * 2);
  unsigned short* headWt = (unsigned short*)alloc((size_t)D * V * 2);
  float* qbuf = (float*)alloc((size_t)16 * D * 4);
  float* attp = (float*)alloc((size_t)16 * D * 4);
  float* ffp  = (float*)alloc((size_t)16 * D * 4);
  float* macc = (float*)alloc((size_t)16 * D * 4);
  float* sbuf = (float*)alloc((size_t)B * H * S * 4);
  unsigned short* attbh  = (unsigned short*)alloc((size_t)16 * D * 2);
  unsigned short* ffbh   = (unsigned short*)alloc((size_t)16 * FF * 2);
  unsigned short* maccbh = (unsigned short*)alloc((size_t)16 * D * 2);

  auto cvt = [&](const float* W, unsigned short* Wtp, int K, int N) {
    long long total = (long long)K * N;
    int blocks = (int)((total + 255) / 256);
    k_convert_transpose<<<blocks, 256, 0, stream>>>(W, Wtp, K, N);
  };
  cvt(kW, kWt, D, D);
  cvt(vW, vWt, D, D);
  for (int l = 0; l < L; ++l) cvt(qW  + (size_t)l * D * D,  qWt  + (size_t)l * D * D,  D,  D);
  for (int l = 0; l < L; ++l) cvt(oW  + (size_t)l * D * D,  oWt  + (size_t)l * D * D,  D,  D);
  for (int l = 0; l < L; ++l) cvt(f1W + (size_t)l * D * FF, f1Wt + (size_t)l * D * FF, D,  FF);
  for (int l = 0; l < L; ++l) cvt(f2W + (size_t)l * FF * D, f2Wt + (size_t)l * FF * D, FF, D);
  cvt(headW, headWt, D, V);

  k_embed<<<B * S, 256, 0, stream>>>(ids, tok, pos, emb);

  SeqArgs a;
  a.emb = emb; a.lw = lw;
  a.kb = kb; a.vb = vb; a.qb = qb; a.ob = ob; a.f1b = f1b; a.f2b = f2b;
  a.ln1w = ln1w; a.ln1b = ln1b; a.ln2w = ln2w; a.ln2b = ln2b;
  a.kWt = kWt; a.vWt = vWt; a.qWt = qWt; a.oWt = oWt; a.f1Wt = f1Wt; a.f2Wt = f2Wt;
  a.memk = memk; a.memv = memv; a.hidden = hidden;
  a.qbuf = qbuf; a.attp = attp; a.ffp = ffp; a.macc = macc; a.sbuf = sbuf;
  a.attbh = attbh; a.ffbh = ffbh; a.maccbh = maccbh;
  k_seq<<<1, 1024, 0, stream>>>(a);

  k_lnf<<<B * S, 256, 0, stream>>>(hidden, lnfw, lnfb, hl);
  k_head<<<(B * S / 16) * (V / 16) / 8, 256, 0, stream>>>(hl, headWt, out);
}
